// GroupChatModel_18803366822319
// MI455X (gfx1250) — compile-verified, matrix operand order checked by reading, not run on hardware
//
#include <hip/hip_runtime.h>
#include <hip/hip_bf16.h>
#include <math.h>

typedef __attribute__((ext_vector_type(16))) _Float16 v16h;
typedef __attribute__((ext_vector_type(8)))  float    v8f;
typedef __attribute__((ext_vector_type(4)))  unsigned int u32x4;
typedef __attribute__((ext_vector_type(4)))  int i32x4;
typedef __attribute__((ext_vector_type(8)))  int i32x8;

static constexpr int kH  = 512;    // hidden / embedding dim
static constexpr int k3H = 1536;   // 3*H (GRU gates)
static constexpr int kB  = 64;     // batch
static constexpr int kS  = 15;     // context utterances
static constexpr int kT  = 30;     // tokens per utterance
static constexpr int kR  = 2;      // responses
static constexpr int kNU = kS * kB;   // 960  fused utter batch
static constexpr int kNR = kR * kB;   // 128  fused resp batch

// ---- Tensor Data Mover availability (device pass on gfx1250 toolchains) ----
#if defined(__has_builtin)
#  if __has_builtin(__builtin_amdgcn_tensor_load_to_lds)
#    define GEMM_USE_TDM 1
#  endif
#endif
#ifndef GEMM_USE_TDM
#  define GEMM_USE_TDM 0
#endif
#if GEMM_USE_TDM
#  if __has_include(<hip/amd_detail/amd_gfx1250_TDM.h>)
#    define TDM_6ARG 1            // therock-10.0 headers -> 6-arg builtin
#  else
#    define TDM_6ARG 0            // ROCm 7.2 -> 5-arg builtin
#  endif
#endif

#define AS3 __attribute__((address_space(3)))

__device__ __forceinline__ float sigmoidf_(float x) { return 1.0f / (1.0f + __expf(-x)); }

#if GEMM_USE_TDM
// DMA `rows` consecutive rows of a (N,K) f16 matrix (row-major, K=512) into LDS.
// 2-D descriptor: data_size=2B, tile_dim0=K units, tile_dim1=rows, dim0 stride=K.
__device__ __forceinline__ void tdm_load_rows_f16(const _Float16* gsrc,
                                                  unsigned lds_byte_off,
                                                  int rows, int K)
{
    unsigned long long ga = (unsigned long long)(size_t)gsrc;
    u32x4 g0;
    g0.x = 1u;                                          // count=1, user mode, no gather
    g0.y = lds_byte_off;                                // D#.lds_addr (bytes)
    g0.z = (unsigned)(ga & 0xFFFFFFFFu);                // global_addr[31:0]
    g0.w = (unsigned)((ga >> 32) & 0x01FFFFFFu)         // global_addr[56:32]
         | (2u << 30);                                  // type = 2 ("image")
    i32x8 g1;
    g1[0] = (int)(1u << 16);                            // data_size=1 -> 2 bytes
    g1[1] = (int)((unsigned)(K & 0xFFFF) << 16);        // tensor_dim0[15:0]
    g1[2] = (int)(((unsigned)K >> 16) & 0xFFFFu)        // tensor_dim0[31:16]
          | (int)((unsigned)(rows & 0xFFFF) << 16);     // tensor_dim1[15:0]
    g1[3] = (int)(((unsigned)rows >> 16) & 0xFFFFu)     // tensor_dim1[31:16]
          | (int)((unsigned)(K & 0xFFFF) << 16);        // tile_dim0 = K
    g1[4] = (int)(rows & 0xFFFF);                       // tile_dim1 = rows, tile_dim2 = 0
    g1[5] = K;                                          // tensor_dim0_stride[31:0]
    g1[6] = 0;                                          // stride0 hi | stride1 lo
    g1[7] = 0;                                          // stride1 hi
    i32x4 g2 = {0, 0, 0, 0};
    i32x4 g3 = {0, 0, 0, 0};
#if TDM_6ARG
    i32x8 g4 = {0, 0, 0, 0, 0, 0, 0, 0};
    __builtin_amdgcn_tensor_load_to_lds(g0, g1, g2, g3, g4, 0);
#else
    __builtin_amdgcn_tensor_load_to_lds(g0, g1, g2, g3, 0);
#endif
}
#endif

// ---------------------------------------------------------------------------
// Generic WMMA GEMM:  C[M,N] = rowgather(A) * W^T (+ bias)   or  C += ... (acc)
//   A : fp32, row m is A[gather[m]] (if gather) else A[m], leading dim K
//   W : f16, (N,K) row-major (we always compute x @ W^T)
//   K = 512 here; M, N multiples of 64.
// Block: 128 threads = 4 waves; block tile 64x64, wave tile 16x64.
// Entire 64(N) x 512(K) weight slab (64 KB) DMA'd into LDS once per block via
// TDM (one 16-row descriptor per wave), overlapped with the C-fragment init;
// the K-loop then runs barrier-free out of LDS.
// ---------------------------------------------------------------------------
__global__ __launch_bounds__(128)
void gemm_f16_wmma(const float* __restrict__ A, const int* __restrict__ gather,
                   const _Float16* __restrict__ W, const float* __restrict__ bias,
                   float* __restrict__ C, int M, int N, int K, int acc)
{
    __shared__ _Float16 Bs[64 * 512];   // 64 KB weight slab

    const int tid  = threadIdx.x;
    const int wave = tid >> 5;
    const int lane = tid & 31;
    const int l16  = lane & 15;
    const int hi   = (lane >> 4) & 1;    // lane group: 0 = lanes 0-15, 1 = 16-31

    const int bm = blockIdx.x * 64;
    const int bn = blockIdx.y * 64;

#if GEMM_USE_TDM
    {   // each wave DMAs its own 16-row slab of W into LDS
        const _Float16* gsrc = W + (size_t)(bn + wave * 16) * K;
        unsigned lds_base = (unsigned)(size_t)(AS3 const void*)Bs;
        tdm_load_rows_f16(gsrc, lds_base + (unsigned)(wave * 16 * K * 2), 16, K);
    }
#else
    {   // fallback: cooperative staging of the whole slab
        for (int idx = tid * 8; idx < 64 * 512; idx += 128 * 8) {
            const _Float16* src = W + (size_t)bn * K + idx;
            #pragma unroll
            for (int e = 0; e < 8; ++e) Bs[idx + e] = src[e];
        }
    }
#endif

    // A fragment source row (both lane halves carry the same M row)
    const int am = bm + wave * 16 + l16;
    const float* arow = gather ? (A + (size_t)gather[am] * K)
                               : (A + (size_t)am * K);

    // C fragment rows: lanes 0-15 -> m = base+v ; lanes 16-31 -> m = base+8+v
    const int cm = bm + wave * 16 + hi * 8;

    v8f cacc[4];   // init overlaps with the weight DMA
    for (int j = 0; j < 4; ++j) {
        const int n = bn + j * 16 + l16;
        if (acc) {
            for (int v = 0; v < 8; ++v)
                cacc[j][v] = C[(size_t)(cm + v) * N + n];
        } else {
            const float bv = bias ? bias[n] : 0.0f;
            for (int v = 0; v < 8; ++v) cacc[j][v] = bv;
        }
    }

#if GEMM_USE_TDM
    __builtin_amdgcn_s_wait_tensorcnt(0);
#endif
    __syncthreads();   // slab resident for all 4 waves

    const int kshift = hi * 8;   // K-half offset per CDNA5 16-bit A layout

    for (int kc = 0; kc < K; kc += 32) {
        v16h af;
        #pragma unroll
        for (int e = 0; e < 16; ++e) {
            const int k = (e < 8 ? e : 8 + e) + kshift;   // K packing per ISA table
            af[e] = (_Float16)arow[kc + k];
        }

        #pragma unroll
        for (int j = 0; j < 4; ++j) {
            v16h bf;
            const _Float16* bsrc = Bs + (size_t)(j * 16 + l16) * K + kc;
            #pragma unroll
            for (int e = 0; e < 16; ++e) {
                const int k = (e < 8 ? e : 8 + e) + kshift;
                bf[e] = bsrc[k];
            }
            cacc[j] = __builtin_amdgcn_wmma_f32_16x16x32_f16(
                false, af, false, bf, (short)0, cacc[j], false, false);
        }
    }

    for (int j = 0; j < 4; ++j) {
        const int n = bn + j * 16 + l16;
        #pragma unroll
        for (int v = 0; v < 8; ++v)
            C[(size_t)(cm + v) * N + n] = cacc[j][v];
    }
}

// fp32 -> f16 weight conversion, optionally transposed (dst becomes (cols,rows))
__global__ void convert_f16_k(const float* __restrict__ src, _Float16* __restrict__ dst,
                              int rows, int cols, int transpose)
{
    int i = blockIdx.x * blockDim.x + threadIdx.x;
    if (i >= rows * cols) return;
    int r = i / cols, c = i % cols;
    _Float16 v = (_Float16)src[i];
    if (transpose) dst[(size_t)c * rows + r] = v;
    else           dst[i] = v;
}

// Per-timestep token index tables (gather rows for the embedding GEMM)
//   tok_u[t*960 + s*64 + b] = context[b,s,t]
//   tok_r[t*128 + r*64 + b] = response[b,r,t]
__global__ void build_tok_k(const int* __restrict__ context, const int* __restrict__ response,
                            int* __restrict__ tok_u, int* __restrict__ tok_r)
{
    int i = blockIdx.x * blockDim.x + threadIdx.x;
    const int NU = kT * kNU, NR = kT * kNR;
    if (i < NU) {
        int t = i / kNU; int rem = i % kNU; int s = rem / kB; int b = rem % kB;
        tok_u[i] = context[((size_t)b * kS + s) * kT + t];
    } else if (i < NU + NR) {
        int j = i - NU;
        int t = j / kNR; int rem = j % kNR; int r = rem / kB; int b = rem % kB;
        tok_r[j] = response[((size_t)b * kR + r) * kT + t];
    }
}

// GRU gate fusion: h <- (1-z)*n + z*h ; optionally record the step output
__global__ void gru_gate_k(const float* __restrict__ xg, const float* __restrict__ hg,
                           float* __restrict__ h, float* __restrict__ out_seq, int M)
{
    int i = blockIdx.x * blockDim.x + threadIdx.x;
    if (i >= M * kH) return;
    int m = i / kH, j = i % kH;
    const float* x = xg + (size_t)m * k3H;
    const float* g = hg + (size_t)m * k3H;
    float r = sigmoidf_(x[j]        + g[j]);
    float z = sigmoidf_(x[kH + j]   + g[kH + j]);
    float n = tanhf    (x[2*kH + j] + r * g[2*kH + j]);
    float hn = (1.0f - z) * n + z * h[i];
    h[i] = hn;
    if (out_seq) out_seq[i] = hn;
}

// logits[r,b,s] = sum_k tanh(Wctx[s,b,k] + wr[r*64+b,k]) * w_alpha[k]  (wave per (r,b,s))
__global__ __launch_bounds__(128)
void attn_logits_k(const float* __restrict__ Wctx, const float* __restrict__ wr,
                   const float* __restrict__ walpha, float* __restrict__ logits)
{
    int gw   = (blockIdx.x * blockDim.x + threadIdx.x) >> 5;
    int lane = threadIdx.x & 31;
    if (gw >= kR * kB * kS) return;
    int r = gw / (kB * kS); int rem = gw % (kB * kS); int b = rem / kS; int s = rem % kS;
    const float* wc = Wctx + ((size_t)s * kB + b) * kH;
    const float* wv = wr   + ((size_t)(r * kB + b)) * kH;
    float acc = 0.0f;
    for (int k = lane; k < kH; k += 32)
        acc += tanhf(wc[k] + wv[k]) * walpha[k];
    for (int o = 16; o > 0; o >>= 1) acc += __shfl_xor(acc, o, 32);
    if (lane == 0) logits[(size_t)(r * kB + b) * kS + s] = acc;
}

// softmax over s (15) for each of 128 (r,b) rows
__global__ void softmax_k(const float* __restrict__ logits, float* __restrict__ alpha)
{
    int i = blockIdx.x * blockDim.x + threadIdx.x;
    if (i >= kR * kB) return;
    const float* row = logits + (size_t)i * kS;
    float mx = row[0];
    for (int s = 1; s < kS; ++s) mx = fmaxf(mx, row[s]);
    float e[kS]; float sum = 0.0f;
    for (int s = 0; s < kS; ++s) { e[s] = __expf(row[s] - mx); sum += e[s]; }
    float inv = 1.0f / sum;
    for (int s = 0; s < kS; ++s) alpha[(size_t)i * kS + s] = e[s] * inv;
}

// ratt[rb,h] = sum_s alpha[rb,s]*ctx_out[s,b,h] + tanh(lin[rb,h])
__global__ void attn_combine_k(const float* __restrict__ alpha, const float* __restrict__ ctx_out,
                               const float* __restrict__ lin, float* __restrict__ ratt)
{
    int i = blockIdx.x * blockDim.x + threadIdx.x;
    if (i >= kR * kB * kH) return;
    int rb = i / kH, hcol = i % kH;
    int b = rb % kB;
    float att = 0.0f;
    for (int s = 0; s < kS; ++s)
        att += alpha[(size_t)rb * kS + s] * ctx_out[((size_t)s * kB + b) * kH + hcol];
    ratt[i] = att + tanhf(lin[i]);
}

// score[r*64+b] = sum_k cs[b,k] * tanh(star[r*64+b,k])   (wave per (r,b))
__global__ __launch_bounds__(128)
void score_k(const float* __restrict__ cs, const float* __restrict__ star,
             float* __restrict__ out)
{
    int gw   = (blockIdx.x * blockDim.x + threadIdx.x) >> 5;
    int lane = threadIdx.x & 31;
    if (gw >= kR * kB) return;
    int b = gw % kB;
    float acc = 0.0f;
    for (int k = lane; k < kH; k += 32)
        acc += cs[(size_t)b * kH + k] * tanhf(star[(size_t)gw * kH + k]);
    for (int o = 16; o > 0; o >>= 1) acc += __shfl_xor(acc, o, 32);
    if (lane == 0) out[gw] = acc;
}

// ---------------------------------------------------------------------------
extern "C" void kernel_launch(void* const* d_in, const int* in_sizes, int n_in,
                              void* d_out, int out_size, void* d_ws, size_t ws_size,
                              hipStream_t stream)
{
    (void)in_sizes; (void)n_in; (void)out_size; (void)ws_size;

    const int*   context  = (const int*)  d_in[0];
    const int*   response = (const int*)  d_in[1];
    const float* emb      = (const float*)d_in[2];
    const float* uWih = (const float*)d_in[3];
    const float* uWhh = (const float*)d_in[4];
    const float* ubih = (const float*)d_in[5];
    const float* ubhh = (const float*)d_in[6];
    const float* cWih = (const float*)d_in[7];
    const float* cWhh = (const float*)d_in[8];
    const float* cbih = (const float*)d_in[9];
    const float* cbhh = (const float*)d_in[10];
    const float* rWih = (const float*)d_in[11];
    const float* rWhh = (const float*)d_in[12];
    const float* rbih = (const float*)d_in[13];
    const float* rbhh = (const float*)d_in[14];
    const float* Wlin = (const float*)d_in[15];
    const float* Wp   = (const float*)d_in[16];
    const float* Wx   = (const float*)d_in[17];
    const float* Wc   = (const float*)d_in[18];
    const float* Wt   = (const float*)d_in[19];
    const float* Wa   = (const float*)d_in[20];
    const float* Walpha = (const float*)d_in[21];
    const float* Ws   = (const float*)d_in[22];
    float* out = (float*)d_out;

    char* p = (char*)d_ws;
    auto alloc = [&](size_t bytes) -> void* {
        void* q = (void*)p; p += (bytes + 255) & ~(size_t)255; return q;
    };

    // f16 weights (right-multiplied matrices stored transposed -> always A @ W^T)
    _Float16* hWih_u = (_Float16*)alloc((size_t)k3H * kH * 2);
    _Float16* hWhh_u = (_Float16*)alloc((size_t)k3H * kH * 2);
    _Float16* hWih_c = (_Float16*)alloc((size_t)k3H * kH * 2);
    _Float16* hWhh_c = (_Float16*)alloc((size_t)k3H * kH * 2);
    _Float16* hWih_r = (_Float16*)alloc((size_t)k3H * kH * 2);
    _Float16* hWhh_r = (_Float16*)alloc((size_t)k3H * kH * 2);
    _Float16* hWlin  = (_Float16*)alloc((size_t)kH * kH * 2);
    _Float16* hWp    = (_Float16*)alloc((size_t)kH * kH * 2);
    _Float16* hWx    = (_Float16*)alloc((size_t)kH * kH * 2);
    _Float16* hWcT   = (_Float16*)alloc((size_t)kH * kH * 2);
    _Float16* hWtT   = (_Float16*)alloc((size_t)kH * kH * 2);
    _Float16* hWaT   = (_Float16*)alloc((size_t)kH * kH * 2);
    _Float16* hWsT   = (_Float16*)alloc((size_t)kH * kH * 2);

    int*   tok_u   = (int*)  alloc((size_t)kT * kNU * 4);
    int*   tok_r   = (int*)  alloc((size_t)kT * kNR * 4);
    float* h_u     = (float*)alloc((size_t)kNU * kH * 4);
    float* h_c     = (float*)alloc((size_t)kB  * kH * 4);
    float* h_r     = (float*)alloc((size_t)kNR * kH * 4);
    float* xg      = (float*)alloc((size_t)kNU * k3H * 4);
    float* hg      = (float*)alloc((size_t)kNU * k3H * 4);
    float* ctx_out = (float*)alloc((size_t)kS * kB * kH * 4);
    float* resp_out= (float*)alloc((size_t)kT * kNR * kH * 4);
    float* Wctx    = (float*)alloc((size_t)kS * kB * kH * 4);
    float* wr      = (float*)alloc((size_t)kNR * kH * 4);
    float* lin     = (float*)alloc((size_t)kNR * kH * 4);
    float* ratt    = (float*)alloc((size_t)kNR * kH * 4);
    float* logits  = (float*)alloc((size_t)kR * kB * kS * 4);
    float* alpha   = (float*)alloc((size_t)kR * kB * kS * 4);
    float* star    = (float*)alloc((size_t)kNR * kH * 4);
    float* cs      = (float*)alloc((size_t)kB * kH * 4);

    auto cvt = [&](const float* s, _Float16* d, int rows, int cols, int tr) {
        int n = rows * cols;
        convert_f16_k<<<dim3((n + 255) / 256), dim3(256), 0, stream>>>(s, d, rows, cols, tr);
    };
    auto gemm = [&](const float* A, const int* gth, const _Float16* Wm, const float* bias,
                    float* Cm, int M, int N, int acc) {
        gemm_f16_wmma<<<dim3(M / 64, N / 64), dim3(128), 0, stream>>>(
            A, gth, Wm, bias, Cm, M, N, kH, acc);
    };
    auto gate = [&](float* hbuf, float* seq, int M) {
        gru_gate_k<<<dim3((M * kH + 255) / 256), dim3(256), 0, stream>>>(xg, hg, hbuf, seq, M);
    };

    // --- weight conversion ---
    cvt(uWih, hWih_u, k3H, kH, 0);  cvt(uWhh, hWhh_u, k3H, kH, 0);
    cvt(cWih, hWih_c, k3H, kH, 0);  cvt(cWhh, hWhh_c, k3H, kH, 0);
    cvt(rWih, hWih_r, k3H, kH, 0);  cvt(rWhh, hWhh_r, k3H, kH, 0);
    cvt(Wlin, hWlin, kH, kH, 0);    cvt(Wp, hWp, kH, kH, 0);   cvt(Wx, hWx, kH, kH, 0);
    cvt(Wc, hWcT, kH, kH, 1);       cvt(Wt, hWtT, kH, kH, 1);
    cvt(Wa, hWaT, kH, kH, 1);       cvt(Ws, hWsT, kH, kH, 1);

    // --- token gather tables ---
    {
        int tot = kT * kNU + kT * kNR;
        build_tok_k<<<dim3((tot + 255) / 256), dim3(256), 0, stream>>>(context, response, tok_u, tok_r);
    }

    // --- utter GRU: N=960 fused (s,b) batch, 30 steps ---
    hipMemsetAsync(h_u, 0, (size_t)kNU * kH * 4, stream);
    for (int t = 0; t < kT; ++t) {
        gemm(emb, tok_u + (size_t)t * kNU, hWih_u, ubih, xg, kNU, k3H, 0);
        gemm(h_u, nullptr,                 hWhh_u, ubhh, hg, kNU, k3H, 0);
        gate(h_u, nullptr, kNU);
    }
    // h_u now holds utter_hidden laid out as (s,b,H)

    // --- ctx GRU: batch 64, 15 steps over utter hiddens ---
    hipMemsetAsync(h_c, 0, (size_t)kB * kH * 4, stream);
    for (int s = 0; s < kS; ++s) {
        gemm(h_u + (size_t)s * kB * kH, nullptr, hWih_c, cbih, xg, kB, k3H, 0);
        gemm(h_c, nullptr,                       hWhh_c, cbhh, hg, kB, k3H, 0);
        gate(h_c, ctx_out + (size_t)s * kB * kH, kB);
    }
    // h_c = context_hidden ; ctx_out = context_output (s,b,H)

    // --- resp GRU: N=128 fused (r,b) batch, 30 steps ---
    hipMemsetAsync(h_r, 0, (size_t)kNR * kH * 4, stream);
    for (int t = 0; t < kT; ++t) {
        gemm(emb, tok_r + (size_t)t * kNR, hWih_r, rbih, xg, kNR, k3H, 0);
        gemm(h_r, nullptr,                 hWhh_r, rbhh, hg, kNR, k3H, 0);
        gate(h_r, resp_out + (size_t)t * kNR * kH, kNR);
    }
    // h_r = resp_hidden ; resp_out[t] rows = (r*64+b)

    // --- Wctx = context_output @ W_c  (stored (s,b,k)) ---
    gemm(ctx_out, nullptr, hWcT, nullptr, Wctx, kS * kB, kH, 0);

    // --- attention scan, both responses fused (rows r*64+b), 30 steps ---
    hipMemsetAsync(ratt, 0, (size_t)kNR * kH * 4, stream);
    for (int t = 0; t < kT; ++t) {
        gemm(resp_out + (size_t)t * kNR * kH, nullptr, hWtT, nullptr, wr, kNR, kH, 0);
        gemm(ratt, nullptr, hWaT, nullptr, wr, kNR, kH, 1);      // wr += ratt @ W_a
        gemm(ratt, nullptr, hWlin, nullptr, lin, kNR, kH, 0);    // lin = ratt @ W_lin^T
        attn_logits_k<<<dim3((kR * kB * kS * 32 + 127) / 128), dim3(128), 0, stream>>>(
            Wctx, wr, Walpha, logits);
        softmax_k<<<dim3(1), dim3(128), 0, stream>>>(logits, alpha);
        attn_combine_k<<<dim3((kR * kB * kH + 255) / 256), dim3(256), 0, stream>>>(
            alpha, ctx_out, lin, ratt);
    }
    // ratt = resp_atten (rows r*64+b)

    // --- final scoring ---
    gemm(ratt, nullptr, hWp, nullptr, star, kNR, kH, 0);   // resp_atten @ W_p^T
    gemm(h_r,  nullptr, hWx, nullptr, star, kNR, kH, 1);   // += resp_hidden @ W_x^T
    gemm(h_c,  nullptr, hWsT, nullptr, cs, kB, kH, 0);     // context_hidden @ W_s
    score_k<<<dim3((kR * kB * 32 + 127) / 128), dim3(128), 0, stream>>>(cs, star, out);
}
